// ParallelMlp_40072044871840
// MI455X (gfx1250) — matrix operand order repaired and checked by reading, not compiled, not run on hardware
//
#include <hip/hip_runtime.h>

// ---------------- types for WMMA ----------------
typedef __attribute__((ext_vector_type(16))) __bf16 v16bf;
typedef __attribute__((ext_vector_type(8)))  float  v8f;

union Frag {
    v16bf v;
    uint4 q[2];
};

// f32 -> bf16 round-to-nearest-even
__device__ __forceinline__ unsigned short f2bf(float f) {
    unsigned int u = __float_as_uint(f);
    u += 0x7fffu + ((u >> 16) & 1u);
    return (unsigned short)(u >> 16);
}

// ---------------- constants ----------------
#define BATCH 8192
#define NHEAD 16
#define DIN   256
#define DHID  512
#define DOUT  256
#define TM    64          // batch rows per workgroup
#define AST   520         // LDS row stride in bf16 elements (512 + 8 pad -> 4-bank skew)
// 2 activation buffers + 2 weight panel buffers (double-buffered async staging)
#define SMEM_BYTES (4 * TM * AST * 2)

// ---------------- async global->LDS staging (CDNA5 ASYNCcnt path) ----------------
// Builtin signature (from compiler diagnostic): pointee type is int4
// (int __attribute__((vector_size(16)))), global side in AS1, LDS side in AS3.
typedef int v4i __attribute__((vector_size(16)));

#if __has_builtin(__builtin_amdgcn_global_load_async_to_lds_b128)
#define ASYNC_STAGE 1
__device__ __forceinline__ void async_cp16(const unsigned short* g, unsigned short* l) {
    // generic->AS1 / AS3 via integer value (low 32 bits of a generic LDS addr == LDS offset)
    __builtin_amdgcn_global_load_async_to_lds_b128(
        (__attribute__((address_space(1))) v4i*)(uintptr_t)g,
        (__attribute__((address_space(3))) v4i*)(unsigned int)(uintptr_t)l,
        0, 0);
}
#else
#define ASYNC_STAGE 0
#endif

__device__ __forceinline__ void wait_async() {
#if ASYNC_STAGE
#if __has_builtin(__builtin_amdgcn_s_wait_asynccnt)
    __builtin_amdgcn_s_wait_asynccnt(0);
#else
    asm volatile("s_wait_asynccnt 0x0" ::: "memory");
#endif
#endif
}

// ---------------- f32 -> bf16 conversion pre-pass ----------------
__global__ __launch_bounds__(256) void cvt_f32_bf16(const float4* __restrict__ src,
                                                    uint2* __restrict__ dst, int n4) {
    int i = blockIdx.x * 256 + threadIdx.x;
    int stride = gridDim.x * 256;
    for (; i < n4; i += stride) {
        float4 v = src[i];
        uint2 o;
        o.x = (unsigned)f2bf(v.x) | ((unsigned)f2bf(v.y) << 16);
        o.y = (unsigned)f2bf(v.z) | ((unsigned)f2bf(v.w) << 16);
        dst[i] = o;
    }
}

// ---------------- weight panel stage: 64 rows x K bf16 -> LDS ----------------
template <int K>
__device__ __forceinline__ void stage_panel(const unsigned short* __restrict__ wp,
                                            unsigned short* __restrict__ sWbuf, int tid) {
    constexpr int CPR = K / 8;  // uint4 chunks per row
    for (int f = tid; f < 64 * CPR; f += 256) {
        int r = f / CPR;
        int c = f - r * CPR;
#if ASYNC_STAGE
        async_cp16(wp + r * K + c * 8, sWbuf + r * AST + c * 8);
#else
        *(uint4*)(sWbuf + r * AST + c * 8) = *(const uint4*)(wp + r * K + c * 8);
#endif
    }
}

// ---------------- fused layer (templated on K, N) ----------------
// out[m,n] = act(in[m,:] . W[n,:] + b[n]) for a 64-row tile; weight panels are
// double-buffered: panel p+1 streams to LDS (async) while panel p computes.
template <int K, int N, bool RELU, bool GOUT>
__device__ __forceinline__ void mlp_layer(const unsigned short* __restrict__ sIn,
                                          unsigned short* __restrict__ sOut,
                                          unsigned short* __restrict__ sW0,
                                          unsigned short* __restrict__ sW1,
                                          const unsigned short* __restrict__ wHead,
                                          const float* __restrict__ bHead,
                                          float* __restrict__ gOut,  // d_out, layer 3 only
                                          int row0, int h,
                                          int tid, int ln, int g, int rowBlk, int colGrp) {
    constexpr int PANELS = N / 64;  // 64 output columns per panel

    const unsigned short* aBase = sIn + (rowBlk * 16 + ln) * AST + g * 8;

    // prologue: stage panel 0 (also flushes any async ops issued by the caller)
    stage_panel<K>(wHead, sW0, tid);
    wait_async();
    __syncthreads();

    for (int p = 0; p < PANELS; ++p) {
        const unsigned short* sWc = (p & 1) ? sW1 : sW0;
        // kick off next panel into the other buffer (its previous consumers
        // finished before the barrier that ended iteration p-1)
        if (p + 1 < PANELS)
            stage_panel<K>(wHead + (size_t)(p + 1) * 64 * K, (p & 1) ? sW0 : sW1, tid);

        // ---- compute: this wave owns 16 rows x 2 column tiles of this panel ----
        v8f acc0 = {0.f, 0.f, 0.f, 0.f, 0.f, 0.f, 0.f, 0.f};
        v8f acc1 = {0.f, 0.f, 0.f, 0.f, 0.f, 0.f, 0.f, 0.f};
        const unsigned short* bBase0 = sWc + (colGrp * 32 + ln) * AST + g * 8;
        const unsigned short* bBase1 = bBase0 + 16 * AST;

#pragma unroll
        for (int k = 0; k < K; k += 32) {
            Frag a, b0, b1;
            // A 16x32 bf16: lane ln holds row m, K in {8g..8g+7} U {16+8g..23+8g}
            a.q[0] = *(const uint4*)(aBase + k);
            a.q[1] = *(const uint4*)(aBase + k + 16);
            // B 32x16 bf16: lane ln holds col n, K in {8g..8g+7} U {16+8g..23+8g}
            b0.q[0] = *(const uint4*)(bBase0 + k);
            b0.q[1] = *(const uint4*)(bBase0 + k + 16);
            b1.q[0] = *(const uint4*)(bBase1 + k);
            b1.q[1] = *(const uint4*)(bBase1 + k + 16);
            acc0 = __builtin_amdgcn_wmma_f32_16x16x32_bf16(false, a.v, false, b0.v,
                                                           (short)0, acc0, false, false);
            acc1 = __builtin_amdgcn_wmma_f32_16x16x32_bf16(false, a.v, false, b1.v,
                                                           (short)0, acc1, false, false);
        }

        // ---- epilogue: bias (+ReLU), write bf16 to LDS or f32 to global ----
        const int n0 = p * 64 + colGrp * 32 + ln;
        const float bias0 = bHead[n0];
        const float bias1 = bHead[n0 + 16];
#pragma unroll
        for (int r = 0; r < 8; ++r) {
            int m = rowBlk * 16 + r + 8 * g;  // D layout: vgpr r -> M = r + 8*(lane/16)
            float v0 = acc0[r] + bias0;
            float v1 = acc1[r] + bias1;
            if (RELU) {
                v0 = v0 > 0.f ? v0 : 0.f;
                v1 = v1 > 0.f ? v1 : 0.f;
            }
            if (GOUT) {
                // out[b][h][o] flat == reference's reshape quirk memory layout
                float* o = gOut + ((size_t)(row0 + m) * NHEAD + h) * DOUT + n0;
                o[0]  = v0;
                o[16] = v1;
            } else {
                sOut[m * AST + n0]      = f2bf(v0);
                sOut[m * AST + n0 + 16] = f2bf(v1);
            }
        }

        wait_async();      // next panel's async copies have landed
        __syncthreads();   // publish sOut / protect weight buffers
    }
}

// ---------------- main fused MLP kernel ----------------
__global__ __launch_bounds__(256) void mlp_kernel(const unsigned short* __restrict__ xb,
                                                  const unsigned short* __restrict__ w1b,
                                                  const unsigned short* __restrict__ w2b,
                                                  const unsigned short* __restrict__ w3b,
                                                  const float* __restrict__ b1,
                                                  const float* __restrict__ b2,
                                                  const float* __restrict__ b3,
                                                  float* __restrict__ out) {
    extern __shared__ uint4 smem4[];
    unsigned short* sm  = (unsigned short*)smem4;
    unsigned short* sA0 = sm;                 // 64 x 520 bf16 activation buffer A
    unsigned short* sA1 = sm + TM * AST;      // 64 x 520 bf16 activation buffer B
    unsigned short* sW0 = sm + 2 * TM * AST;  // weight panel buffer 0
    unsigned short* sW1 = sm + 3 * TM * AST;  // weight panel buffer 1

    const int tid    = threadIdx.x;
    const int lane   = tid & 31;
    const int wave   = tid >> 5;
    const int rowBlk = wave & 3;   // which 16-row block of the 64-row tile
    const int colGrp = wave >> 2;  // which half of the 64-col weight panel
    const int g      = lane >> 4;
    const int ln     = lane & 15;

    const int h    = blockIdx.x >> 7;          // 16 heads; consecutive blocks share a head
    const int row0 = (blockIdx.x & 127) * TM;  // 128 batch tiles

    // ---- stage x tile (64 x 256 bf16) into sA0; wait folded into layer-1 prologue ----
    {
        const unsigned short* src = xb + (size_t)row0 * DIN;
#pragma unroll
        for (int i = 0; i < 8; ++i) {
            int f = tid + i * 256;  // 2048 16B chunks total; 32 per row
            int r = f >> 5;
            int c = f & 31;
#if ASYNC_STAGE
            async_cp16(src + r * DIN + c * 8, sA0 + r * AST + c * 8);
#else
            *(uint4*)(sA0 + r * AST + c * 8) = *(const uint4*)(src + r * DIN + c * 8);
#endif
        }
    }

    mlp_layer<DIN, DHID, true, false>(sA0, sA1, sW0, sW1, w1b + (size_t)h * DHID * DIN,
                                      b1 + h * DHID, nullptr, 0, 0,
                                      tid, ln, g, rowBlk, colGrp);
    mlp_layer<DHID, DHID, true, false>(sA1, sA0, sW0, sW1, w2b + (size_t)h * DHID * DHID,
                                       b2 + h * DHID, nullptr, 0, 0,
                                       tid, ln, g, rowBlk, colGrp);
    mlp_layer<DHID, DOUT, false, true>(sA0, nullptr, sW0, sW1, w3b + (size_t)h * DOUT * DHID,
                                       b3 + h * DOUT, out, row0, h,
                                       tid, ln, g, rowBlk, colGrp);
}

// ---------------- launch ----------------
extern "C" void kernel_launch(void* const* d_in, const int* in_sizes, int n_in,
                              void* d_out, int out_size, void* d_ws, size_t ws_size,
                              hipStream_t stream) {
    (void)in_sizes; (void)n_in; (void)out_size; (void)ws_size;

    const float* x  = (const float*)d_in[0];
    const float* W1 = (const float*)d_in[1];
    const float* b1 = (const float*)d_in[2];
    const float* W2 = (const float*)d_in[3];
    const float* b2 = (const float*)d_in[4];
    const float* W3 = (const float*)d_in[5];
    const float* b3 = (const float*)d_in[6];

    // bf16 scratch layout (element counts)
    const size_t nX  = (size_t)BATCH * DIN;          // 2,097,152
    const size_t nW1 = (size_t)NHEAD * DHID * DIN;   // 2,097,152
    const size_t nW2 = (size_t)NHEAD * DHID * DHID;  // 4,194,304
    const size_t nW3 = (size_t)NHEAD * DOUT * DHID;  // 2,097,152
    unsigned short* ws  = (unsigned short*)d_ws;
    unsigned short* xb  = ws;
    unsigned short* w1b = xb + nX;
    unsigned short* w2b = w1b + nW1;
    unsigned short* w3b = w2b + nW2;

    cvt_f32_bf16<<<2048, 256, 0, stream>>>((const float4*)x,  (uint2*)xb,  (int)(nX  / 4));
    cvt_f32_bf16<<<2048, 256, 0, stream>>>((const float4*)W1, (uint2*)w1b, (int)(nW1 / 4));
    cvt_f32_bf16<<<2048, 256, 0, stream>>>((const float4*)W2, (uint2*)w2b, (int)(nW2 / 4));
    cvt_f32_bf16<<<2048, 256, 0, stream>>>((const float4*)W3, (uint2*)w3b, (int)(nW3 / 4));

    // allow ~260 KB dynamic LDS (320 KB available per WGP on gfx1250)
    (void)hipFuncSetAttribute((const void*)mlp_kernel,
                              hipFuncAttributeMaxDynamicSharedMemorySize, SMEM_BYTES);

    mlp_kernel<<<NHEAD * (BATCH / TM), 256, SMEM_BYTES, stream>>>(
        xb, w1b, w2b, w3b, b1, b2, b3, (float*)d_out);
}
